// XGNN_lamda_added_53283364274269
// MI455X (gfx1250) — matrix operand order repaired and checked by reading, not compiled, not run on hardware
//
#include <hip/hip_runtime.h>

// ---------------------------------------------------------------------------
// ChebConv GNN forward on gfx1250 (CDNA5, wave32, WMMA).
// fp32 edge-scatter SpMV (atomic/bandwidth bound), bf16 WMMA GEMMs with fp32
// accumulation. GEMM v2: weights register-resident per wave, A tile staged in
// LDS as bf16 and shared by all 8 N-tile waves of the block.
// ---------------------------------------------------------------------------

typedef __attribute__((ext_vector_type(16))) __bf16 v16bf;
typedef __attribute__((ext_vector_type(8)))  __bf16 v8bf;
typedef __attribute__((ext_vector_type(8)))  float  v8f;

#define NNODES 50000
#define NEDGES 800000
#define FDIM   128     // IN_F == HID
#define KCAT   384     // 3 * 128
#define OUTF   64
#define BN_EPS 1e-5f

// ------------------------------- utility -----------------------------------

__global__ void zero_f32(float* __restrict__ p, long n) {
    long i = (long)blockIdx.x * blockDim.x + threadIdx.x;
    if (i < n) p[i] = 0.0f;
}

__global__ void deg_kernel(const int* __restrict__ dst, float* __restrict__ deg, int E) {
    int e = blockIdx.x * blockDim.x + threadIdx.x;
    if (e < E) atomicAdd(&deg[dst[e]], 1.0f);
}

__global__ void dinv_kernel(float* __restrict__ deg, int n) {
    int i = blockIdx.x * blockDim.x + threadIdx.x;
    if (i < n) deg[i] = rsqrtf(fmaxf(deg[i], 1.0f));
}

// Xcat[i, 0:128] = X[i], Xcat[i, 128:384] = 0 (zero T1/T2 accumulators)
__global__ void init_xcat(const float* __restrict__ X, float* __restrict__ xcat, long nf) {
    long t = (long)blockIdx.x * blockDim.x + threadIdx.x;
    if (t >= nf) return;
    long row = t >> 7;
    int  f   = (int)(t & 127);
    long base = row * KCAT;
    xcat[base + f]       = X[t];
    xcat[base + 128 + f] = 0.0f;
    xcat[base + 256 + f] = 0.0f;
}

// Edge-parallel scatter: xcat[dst, doff + c] += xsrc[src, soff + c] * dinv[src]
// One wave per edge (32 lanes x float4 = full 128-feature row).
__global__ void spmv_scatter(const float* __restrict__ xsrc, int sstride, int soff,
                             float* __restrict__ xcat, int doff,
                             const int* __restrict__ src, const int* __restrict__ dst,
                             const float* __restrict__ dinv, int E) {
    long t = (long)blockIdx.x * blockDim.x + threadIdx.x;
    if (t >= (long)E * 32) return;
    int e = (int)(t >> 5);
    int c = (int)(t & 31) << 2;
    int s = src[e];
    int d = dst[e];
    float w = dinv[s];
    const float4 v = *(const float4*)(xsrc + (long)s * sstride + soff + c);
    float* out = xcat + (long)d * KCAT + doff + c;
    atomicAdd(out + 0, v.x * w);
    atomicAdd(out + 1, v.y * w);
    atomicAdd(out + 2, v.z * w);
    atomicAdd(out + 3, v.w * w);
}

// T1 = -dinv[i] * acc      (re_norm = 2/lambda_max = 1  =>  T1 = -Ahat X)
__global__ void finalize_t1(float* __restrict__ xcat, const float* __restrict__ dinv, long nf) {
    long t = (long)blockIdx.x * blockDim.x + threadIdx.x;
    if (t >= nf) return;
    long row = t >> 7;
    int  f   = (int)(t & 127);
    float* p = xcat + row * KCAT + 128 + f;
    *p = -dinv[row] * (*p);
}

// T2 = -2*dinv[i]*acc - X0[i]   (T2 = -2 Ahat T1 - X)
__global__ void finalize_t2(float* __restrict__ xcat, const float* __restrict__ x0,
                            const float* __restrict__ dinv, long nf) {
    long t = (long)blockIdx.x * blockDim.x + threadIdx.x;
    if (t >= nf) return;
    long row = t >> 7;
    int  f   = (int)(t & 127);
    float* p = xcat + row * KCAT + 256 + f;
    *p = -2.0f * dinv[row] * (*p) - x0[row * 128 + f];
}

// ------------------------------ WMMA GEMM ----------------------------------
// C[N, OUTD] = act(A[N, KD] @ W[OUTD, KD]^T + bias) (+ resid)
// Block = OUTD/16 waves; wave w owns N-tile w with its B fragments held in
// VGPRs for the whole kernel. Grid-stride loop over M-tiles: each iteration
// stages the 16xKD fp32 A tile into LDS as bf16 (cooperative, coalesced
// float4 loads), then every wave computes one 16x16 tile with KD/32
// v_wmma_f32_16x16x32_bf16 ops fed by ds_load_b128 fragments.
// Fragment layouts (A/B: contiguous k order chosen here, not the ISA table order; D as ISA):
//   A: lane m = lane&15, hi = lane>>4, elem e -> K = kb + hi*16 + e
//   B: lane n = lane&15, elem e -> K = kb + hi*16 + e   (B[k][n] = W[n][k])
//   D: elem r -> M = r + hi*8, N = lane&15
template<int KD, int OUTD, bool RELU, bool RESID>
__global__ __launch_bounds__(OUTD * 2)
void wmma_gemm(const float* __restrict__ A, const float* __restrict__ W,
               const float* __restrict__ bias, const float* __restrict__ resid,
               float* __restrict__ C, int mtiles) {
    constexpr int NT   = OUTD / 16;  // waves per block = N-tiles
    constexpr int NKB  = KD / 32;    // WMMA k-steps
    constexpr int KDP  = KD + 8;     // padded bf16 row stride (16B-aligned, bank-spread)
    constexpr int NTHR = NT * 32;
    __shared__ __align__(16) __bf16 sA[16 * KDP];

    const int tid  = threadIdx.x;
    const int lane = tid & 31;
    const int wid  = tid >> 5;       // this wave's N-tile
    const int hi   = lane >> 4;
    const int l15  = lane & 15;

    // --- preload B fragments (weights) into registers, once per wave ---
    const float* wrow = W + (long)(wid * 16 + l15) * KD;
    v16bf bfr[NKB];
#pragma unroll
    for (int kb = 0; kb < NKB; ++kb) {
#pragma unroll
        for (int e = 0; e < 16; ++e)
            bfr[kb][e] = (__bf16)wrow[kb * 32 + hi * 16 + e];
    }

    const int   ncol = wid * 16 + l15;
    const float bv   = bias[ncol];

    for (int mt = blockIdx.x; mt < mtiles; mt += gridDim.x) {
        // --- cooperative stage: A[mt*16 .. +15][0..KD) -> bf16 in LDS ---
        const float* atile = A + (long)mt * 16 * KD;
        constexpr int NQ = 16 * KD / 4;            // float4 quads in the tile
        for (int q = tid; q < NQ; q += NTHR) {
            int row = q / (KD / 4);
            int c4  = (q % (KD / 4)) * 4;
            float4 v = *(const float4*)(atile + row * KD + c4);
            __bf16* dp = &sA[row * KDP + c4];
            dp[0] = (__bf16)v.x; dp[1] = (__bf16)v.y;
            dp[2] = (__bf16)v.z; dp[3] = (__bf16)v.w;
        }
        __syncthreads();

        // --- compute 16x16 tile ---
        v8f acc = {};
#pragma unroll
        for (int kb = 0; kb < NKB; ++kb) {
            const __bf16* ap = &sA[l15 * KDP + kb * 32 + hi * 16];
            v8bf lo  = *(const v8bf*)(ap);          // K = kb + hi*16 + 0..7
            v8bf hi8 = *(const v8bf*)(ap + 8);      // K = kb + hi*16 + 8..15
            v16bf af = __builtin_shufflevector(lo, hi8,
                0, 1, 2, 3, 4, 5, 6, 7, 8, 9, 10, 11, 12, 13, 14, 15);
            acc = __builtin_amdgcn_wmma_f32_16x16x32_bf16(
                /*neg_a=*/false, af, /*neg_b=*/false, bfr[kb],
                /*c_mod=*/(short)0, acc, /*reuse_a=*/false, /*reuse_b=*/false);
        }
        __syncthreads();   // LDS consumed before next stage overwrites it

        // --- epilogue ---
#pragma unroll
        for (int r = 0; r < 8; ++r) {
            long row = (long)mt * 16 + r + hi * 8;
            float v = acc[r] + bv;
            if (RELU)  v = fmaxf(v, 0.0f);
            long idx = row * OUTD + ncol;
            if (RESID) v += resid[idx];
            C[idx] = v;
        }
    }
}

// ------------------------------ BatchNorm ----------------------------------

__global__ void bn_stats(const float* __restrict__ X, float* __restrict__ st, int n) {
    int f = threadIdx.x;  // 128 threads: one feature each
    float s = 0.0f, s2 = 0.0f;
    for (int r = blockIdx.x; r < n; r += gridDim.x) {
        float v = X[(long)r * 128 + f];
        s  += v;
        s2 += v * v;
    }
    atomicAdd(&st[f], s);
    atomicAdd(&st[128 + f], s2);
}

__global__ void bn_apply(float* __restrict__ X, const float* __restrict__ st,
                         const float* __restrict__ gamma, const float* __restrict__ beta,
                         long nf, float invn) {
    long t = (long)blockIdx.x * blockDim.x + threadIdx.x;
    if (t >= nf) return;
    int f = (int)(t & 127);
    float mu  = st[f] * invn;
    float var = st[128 + f] * invn - mu * mu;
    X[t] = (X[t] - mu) * rsqrtf(var + BN_EPS) * gamma[f] + beta[f];
}

// ------------------------------ launcher -----------------------------------

extern "C" void kernel_launch(void* const* d_in, const int* in_sizes, int n_in,
                              void* d_out, int out_size, void* d_ws, size_t ws_size,
                              hipStream_t stream) {
    const float* features = (const float*)d_in[0];
    const int*   src      = (const int*)  d_in[1];
    const int*   dst      = (const int*)  d_in[2];
    const float* W1       = (const float*)d_in[3];
    const float* b1       = (const float*)d_in[4];
    const float* W3       = (const float*)d_in[5];
    const float* b3       = (const float*)d_in[6];
    const float* gamma    = (const float*)d_in[7];
    const float* beta     = (const float*)d_in[8];
    const float* Wm1      = (const float*)d_in[9];
    const float* bm1      = (const float*)d_in[10];
    const float* Wm2      = (const float*)d_in[11];
    const float* bm2      = (const float*)d_in[12];
    float* out = (float*)d_out;

    const int  N = NNODES, E = NEDGES;
    const long nf = (long)N * FDIM;

    // Workspace carve-up (256B aligned).
    char* ws = (char*)d_ws;
    size_t off = 0;
    auto carve = [&](size_t bytes) -> float* {
        float* p = (float*)(ws + off);
        off = (off + bytes + 255) & ~(size_t)255;
        return p;
    };
    float* dinv  = carve((size_t)N * 4);            //  0.2 MB (degree -> dinv in place)
    float* xcat  = carve((size_t)N * KCAT * 4);     // 76.8 MB
    float* X1    = carve((size_t)N * FDIM * 4);     // 25.6 MB (conv1 -> BN, residual)
    float* X2    = carve((size_t)N * FDIM * 4);     // 25.6 MB (conv2 out)
    float* stats = carve(1024);                     // 256 floats (sum, sumsq)
    float* H     = xcat;                            // MLP hidden overlays dead xcat

    const int TB = 256;
    const long spmvT = (long)E * 32;
    const int gN    = (int)((N + TB - 1) / TB);
    const int gE    = (int)((E + TB - 1) / TB);
    const int gNF   = (int)((nf + TB - 1) / TB);
    const int gSPMV = (int)((spmvT + TB - 1) / TB);
    const int mtiles = N / 16;   // 3125, exact
    const int gGEMM  = 625;      // grid-stride: 5 M-tiles per block

    // --- degrees -> dinv ---
    zero_f32<<<gN, TB, 0, stream>>>(dinv, N);
    deg_kernel<<<gE, TB, 0, stream>>>(dst, dinv, E);
    dinv_kernel<<<gN, TB, 0, stream>>>(dinv, N);

    // --- ChebConv 1: build [X | T1 | T2] ---
    init_xcat<<<gNF, TB, 0, stream>>>(features, xcat, nf);
    spmv_scatter<<<gSPMV, TB, 0, stream>>>(features, FDIM, 0, xcat, 128, src, dst, dinv, E);
    finalize_t1<<<gNF, TB, 0, stream>>>(xcat, dinv, nf);
    spmv_scatter<<<gSPMV, TB, 0, stream>>>(xcat, KCAT, 128, xcat, 256, src, dst, dinv, E);
    finalize_t2<<<gNF, TB, 0, stream>>>(xcat, features, dinv, nf);

    // X1 = relu(Xcat @ W1^T + b1)
    wmma_gemm<KCAT, FDIM, true, false><<<gGEMM, 256, 0, stream>>>(xcat, W1, b1, nullptr, X1, mtiles);

    // --- BatchNorm (training-mode batch stats, biased var) ---
    zero_f32<<<1, TB, 0, stream>>>(stats, 256);
    bn_stats<<<1024, FDIM, 0, stream>>>(X1, stats, N);
    bn_apply<<<gNF, TB, 0, stream>>>(X1, stats, gamma, beta, nf, 1.0f / (float)N);

    // --- ChebConv 2: build [X1 | T1 | T2] ---
    init_xcat<<<gNF, TB, 0, stream>>>(X1, xcat, nf);
    spmv_scatter<<<gSPMV, TB, 0, stream>>>(X1, FDIM, 0, xcat, 128, src, dst, dinv, E);
    finalize_t1<<<gNF, TB, 0, stream>>>(xcat, dinv, nf);
    spmv_scatter<<<gSPMV, TB, 0, stream>>>(xcat, KCAT, 128, xcat, 256, src, dst, dinv, E);
    finalize_t2<<<gNF, TB, 0, stream>>>(xcat, X1, dinv, nf);

    // X2 = relu(Xcat @ W3^T + b3) + X1   (residual; relu before add, as reference)
    wmma_gemm<KCAT, FDIM, true, true><<<gGEMM, 256, 0, stream>>>(xcat, W3, b3, X1, X2, mtiles);

    // --- MLP ---
    // H = relu(X2 @ Wm1^T + bm1)   (H overlays xcat, which is dead now)
    wmma_gemm<FDIM, FDIM, true, false><<<gGEMM, 256, 0, stream>>>(X2, Wm1, bm1, nullptr, H, mtiles);
    // out = H @ Wm2^T + bm2
    wmma_gemm<FDIM, OUTF, false, false><<<gGEMM, 128, 0, stream>>>(H, Wm2, bm2, nullptr, out, mtiles);

    (void)in_sizes; (void)n_in; (void)out_size; (void)ws_size;
}
